// CRF_52604759441903
// MI455X (gfx1250) — matrix operand order repaired and checked by reading, not compiled, not run on hardware
//
#include <hip/hip_runtime.h>

typedef __attribute__((ext_vector_type(16))) _Float16 v16h;
typedef __attribute__((ext_vector_type(8)))  float    v8f;

#define L_SEQ     512
#define B_BATCH   1024
#define K_ST      64
#define START_I   62
#define STOP_I    63
#define NEG_INF_F (-10000.0f)
#define LOG2E_F   1.44269504088896340736f
#define LN2_F     0.69314718055994530942f

// ---------------------------------------------------------------------------
// Gold (sentence) score: one thread per batch element, T cached in LDS.
// ---------------------------------------------------------------------------
__global__ __launch_bounds__(256) void crf_gold_kernel(
    const float* __restrict__ emit,    // [L,B,K]
    const float* __restrict__ trans,   // [K,K]
    const int*   __restrict__ labels,  // [L,B]
    const float* __restrict__ masks,   // [L,B]
    float*       __restrict__ gold)    // [B]
{
    __shared__ float Tl[K_ST * K_ST];
    for (int i = threadIdx.x; i < K_ST * K_ST; i += blockDim.x)
        Tl[i] = trans[i];
    __syncthreads();

    const int b = blockIdx.x * blockDim.x + threadIdx.x;
    if (b >= B_BATCH) return;

    int   prev  = START_I;
    int   last  = START_I;
    float score = 0.0f;

    for (int t = 0; t < L_SEQ; ++t) {
        const int   lab = labels[t * B_BATCH + b];
        const float mk  = masks[t * B_BATCH + b];
        const float sc  = Tl[lab * K_ST + prev] +
                          emit[(size_t)(t * B_BATCH + b) * K_ST + lab];
        score = fmaf(sc, mk, score);
        if (mk > 0.0f) last = lab;
        prev = lab;
    }
    score += Tl[STOP_I * K_ST + last];
    gold[b] = score;
}

// ---------------------------------------------------------------------------
// Partition: base-2 exp-domain forward scan, 4 waves per 16-batch tile.
// Wave w owns M-tile w (states 16w..16w+15):
//   al[r] holds alpha2 for state j = 16*w + 8*(lane>=16) + r, batch bb+(lane&15)
// Per step:  [partial max -> LDS] bar [global max; P=exp2(al-m) -> LDS] bar
//            [B frags; 2x v_wmma_f32_16x16x32_f16; log2; masked update]
// ---------------------------------------------------------------------------
__global__ __launch_bounds__(128) void crf_partition_wmma(
    const float* __restrict__ emit,    // [L,B,K]
    const float* __restrict__ trans,   // [K,K]
    const float* __restrict__ masks,   // [L,B]
    const float* __restrict__ gold,    // [B]
    float*       __restrict__ out)     // [B]
{
    // P in exact B-fragment order: frag kf, lane l -> halves [(kf*32+l)*16 .. +15]
    __shared__ __align__(32) _Float16 p_lds[2 * 32 * 16];
    __shared__ __align__(16) float    redbuf[2][16][4];  // [lh][batch][wave]

    const int tid  = threadIdx.x;
    const int lane = tid & 31;
    const int w    = tid >> 5;      // wave id == M-tile
    const int lh   = lane >> 4;     // lane half
    const int lm   = lane & 15;
    const int bb   = blockIdx.x * 16;
    const int bl   = bb + lm;       // this lane's batch column

    // ---- resident A fragments for M-tile w: exp2(T*log2e) in f16, A-layout
    // A 16x32 f16: M = lane&15 ; element e -> k = e + 16*(lane>=16)
    v16h aT[2];
    {
        const int j = 16 * w + lm;
#pragma unroll
        for (int kf = 0; kf < 2; ++kf)
#pragma unroll
            for (int e = 0; e < 16; ++e) {
                const int kloc = 32 * kf + e + (lh ? 16 : 0);
                aT[kf][e] = (_Float16)__builtin_amdgcn_exp2f(
                    trans[j * K_ST + kloc] * LOG2E_F);
            }
    }

    // ---- init alpha2 (base-2 log domain) ----
    float al[8];
#pragma unroll
    for (int r = 0; r < 8; ++r) {
        const int j = 16 * w + 8 * lh + r;
        al[r] = (j == START_I) ? 0.0f : (NEG_INF_F * LOG2E_F);
    }

    // P-store address for this lane (contiguous 8 halves -> one ds_store_b128)
    const int k0   = 16 * w + 8 * lh;
    const int pidx = (k0 >> 5) * 512 + (((k0 >> 4) & 1) * 16 + lm) * 16 + (k0 & 15);

    // induction-variable pointers
    const float* ep = emit  + (size_t)bl * K_ST + 16 * w + 8 * lh;
    const float* mp = masks + bl;

    for (int t = 0; t < L_SEQ; ++t) {
        // hoist global loads (overlap with reductions / LDS traffic)
        const float  mk = *mp;                      mp += B_BATCH;
        const float4 e0 = *(const float4*)(ep);
        const float4 e1 = *(const float4*)(ep + 4); ep += B_BATCH * K_ST;

        // per-wave partial max: balanced tree (+ cross-half shuffle)
        float pm = fmaxf(fmaxf(fmaxf(al[0], al[1]), fmaxf(al[2], al[3])),
                         fmaxf(fmaxf(al[4], al[5]), fmaxf(al[6], al[7])));
        pm = fmaxf(pm, __shfl_xor(pm, 16, 32));
        redbuf[lh][lm][w] = pm;
        __syncthreads();

        // global per-batch max over the 4 waves (one b128 LDS load)
        const float4 pm4 = *(const float4*)&redbuf[lh][lm][0];
        const float  m   = fmaxf(fmaxf(pm4.x, pm4.y), fmaxf(pm4.z, pm4.w));

        // P = exp2(alpha2 - m2) -> fragment-ordered LDS (one b128 store)
#pragma unroll
        for (int r = 0; r < 8; ++r)
            p_lds[pidx + r] = (_Float16)__builtin_amdgcn_exp2f(al[r] - m);
        __syncthreads();

        // B fragments (same for all waves) and the two chained WMMAs
        const v16h b0 = *(const v16h*)&p_lds[lane * 16];
        const v16h b1 = *(const v16h*)&p_lds[512 + lane * 16];

        v8f s = {};
        s = __builtin_amdgcn_wmma_f32_16x16x32_f16(
                false, aT[0], false, b0, (short)0, s, false, false);
        s = __builtin_amdgcn_wmma_f32_16x16x32_f16(
                false, aT[1], false, b1, (short)0, s, false, false);

        const float ev[8] = {e0.x, e0.y, e0.z, e0.w, e1.x, e1.y, e1.z, e1.w};
#pragma unroll
        for (int r = 0; r < 8; ++r) {
            // alpha2' = log2(S) + m2 + emit*log2e
            const float an = fmaf(ev[r], LOG2E_F,
                                  __builtin_amdgcn_logf(s[r]) + m);
            al[r] = (mk > 0.0f) ? an : al[r];
        }
    }

    // ---- final logsumexp2(alpha2 + T[STOP,:]*log2e), reduced across waves ----
#pragma unroll
    for (int r = 0; r < 8; ++r)
        al[r] = fmaf(trans[STOP_I * K_ST + 16 * w + 8 * lh + r], LOG2E_F, al[r]);

    float pm = fmaxf(fmaxf(fmaxf(al[0], al[1]), fmaxf(al[2], al[3])),
                     fmaxf(fmaxf(al[4], al[5]), fmaxf(al[6], al[7])));
    pm = fmaxf(pm, __shfl_xor(pm, 16, 32));
    redbuf[lh][lm][w] = pm;
    __syncthreads();

    const float4 pm4 = *(const float4*)&redbuf[lh][lm][0];
    const float  m   = fmaxf(fmaxf(pm4.x, pm4.y), fmaxf(pm4.z, pm4.w));
    __syncthreads();  // redbuf reused for sums below

    float ss = 0.0f;
#pragma unroll
    for (int r = 0; r < 8; ++r) ss += __builtin_amdgcn_exp2f(al[r] - m);
    ss += __shfl_xor(ss, 16, 32);
    redbuf[lh][lm][w] = ss;
    __syncthreads();

    if (w == 0 && lh == 0) {
        const float4 s4   = *(const float4*)&redbuf[0][lm][0];
        const float  stot = (s4.x + s4.y) + (s4.z + s4.w);
        const float  part = (__builtin_amdgcn_logf(stot) + m) * LN2_F;  // back to ln
        out[bl] = part - gold[bl];
    }
}

// ---------------------------------------------------------------------------
extern "C" void kernel_launch(void* const* d_in, const int* in_sizes, int n_in,
                              void* d_out, int out_size, void* d_ws, size_t ws_size,
                              hipStream_t stream) {
    const float* emit   = (const float*)d_in[0];  // (512,1024,64) f32
    const float* trans  = (const float*)d_in[1];  // (64,64)       f32
    const int*   labels = (const int*)d_in[2];    // (512,1024)    i32
    const float* masks  = (const float*)d_in[3];  // (512,1024)    f32
    float*       out    = (float*)d_out;          // (1024,)       f32
    float*       gold   = (float*)d_ws;           // scratch: 1024 f32

    crf_gold_kernel<<<B_BATCH / 256, 256, 0, stream>>>(emit, trans, labels, masks, gold);
    crf_partition_wmma<<<B_BATCH / 16, 128, 0, stream>>>(emit, trans, masks, gold, out);
}
